// DistancePositionalEmbedding_53429393162765
// MI455X (gfx1250) — matrix operand order repaired and checked by reading, not compiled, not run on hardware
//
#include <hip/hip_runtime.h>

typedef __attribute__((ext_vector_type(16))) _Float16 v16h;
typedef __attribute__((ext_vector_type(8)))  float    v8f;
typedef __attribute__((ext_vector_type(4)))  float    f4;

#define T_BLK 512
#define E_DIM 64
#define H_DIM 128

// ---- WMMA fragment index helpers (CDNA5 ISA 7.12.2, wave32) ----
// A-matrix 16x32 f16: lane (half=lane/16, m=lane%16) holds row M=m.
// VGPR v (elements 2v,2v+1): K base = (v<4 ? 2v : 16+2(v-4)) + 8*half.
__device__ __forceinline__ int a_k_of_elem(int e, int half) {
    int vg = e >> 1;
    int base = (vg < 4) ? (2 * vg) : (16 + 2 * (vg - 4));
    return base + 8 * half + (e & 1);
}
// B-matrix 32x16 f16: lane (half, n=lane%16) holds col N=n; element e <-> K = a_k_of_elem(e, half) (same as A).
// C/D 16x16 f32: element r <-> (M = r + 8*half, N = lane%16).

__device__ __forceinline__ float gelu_exact(float x) {
    return 0.5f * x * (1.0f + erff(x * 0.70710678118654752f));
}

// Kernel 1: Y[512][128] = gelu(pos_table @ W1 + b1) @ W2 + b2
// One wave32 per 16-row tile of d. 32 blocks x 32 threads.
__global__ __launch_bounds__(32)
void dpe_build_table(const float* __restrict__ pos,   // [512][64]
                     const float* __restrict__ W1,    // [64][128]
                     const float* __restrict__ b1,    // [128]
                     const float* __restrict__ W2,    // [128][128]
                     const float* __restrict__ b2,    // [128]
                     float* __restrict__ Y)           // [512][128]
{
    const int lane = threadIdx.x & 31;
    const int half = lane >> 4;
    const int m    = lane & 15;
    const int row0 = blockIdx.x * 16;

    __shared__ _Float16 Hs[16 * H_DIM];   // gelu'd hidden tile, row-major

    // ---- A1 fragments: pos_table rows [row0..row0+15], K in two chunks of 32 ----
    v16h a1[2];
#pragma unroll
    for (int c = 0; c < 2; ++c)
#pragma unroll
        for (int e = 0; e < 16; ++e) {
            int k = 32 * c + a_k_of_elem(e, half);
            a1[c][e] = (_Float16)pos[(row0 + m) * E_DIM + k];
        }

    // ---- GEMM1 (16x64 @ 64x128) + bias + exact GELU -> LDS (f16) ----
#pragma unroll
    for (int t = 0; t < 8; ++t) {           // N tiles of 16
        v8f acc = {};
#pragma unroll
        for (int c = 0; c < 2; ++c) {       // K chunks of 32
            v16h bf;
#pragma unroll
            for (int e = 0; e < 16; ++e) {
                int k = 32 * c + a_k_of_elem(e, half);
                bf[e] = (_Float16)W1[k * H_DIM + 16 * t + m];
            }
            acc = __builtin_amdgcn_wmma_f32_16x16x32_f16(
                false, a1[c], false, bf, (short)0, acc, false, false);
        }
        const int col  = 16 * t + m;
        const float bv = b1[col];
#pragma unroll
        for (int r = 0; r < 8; ++r) {
            float x = acc[r] + bv;
            Hs[(r + 8 * half) * H_DIM + col] = (_Float16)gelu_exact(x);
        }
    }
    __syncthreads();

    // ---- A2 fragments from LDS: [16 x 128] in four K-chunks of 32 ----
    v16h a2[4];
#pragma unroll
    for (int c = 0; c < 4; ++c)
#pragma unroll
        for (int e = 0; e < 16; ++e) {
            int k = 32 * c + a_k_of_elem(e, half);
            a2[c][e] = Hs[m * H_DIM + k];
        }

    // ---- GEMM2 (16x128 @ 128x128) + bias -> Y ----
#pragma unroll
    for (int t = 0; t < 8; ++t) {
        v8f acc = {};
#pragma unroll
        for (int c = 0; c < 4; ++c) {
            v16h bf;
#pragma unroll
            for (int e = 0; e < 16; ++e) {
                int k = 32 * c + a_k_of_elem(e, half);
                bf[e] = (_Float16)W2[k * H_DIM + 16 * t + m];
            }
            acc = __builtin_amdgcn_wmma_f32_16x16x32_f16(
                false, a2[c], false, bf, (short)0, acc, false, false);
        }
        const int col  = 16 * t + m;
        const float bv = b2[col];
#pragma unroll
        for (int r = 0; r < 8; ++r) {
            Y[(row0 + r + 8 * half) * H_DIM + col] = acc[r] + bv;
        }
    }
}

// Kernel 2: out[b][i][j][:] = Y[(j<=i) ? i-j : 0][:]
// Pure store-bandwidth: 256 MB of NT b128 stores; Y reads are L2 hits.
// One block per (b,i): 256 threads, thread = (jg=tid/32, c4=tid%32).
__global__ __launch_bounds__(256)
void dpe_scatter(const f4* __restrict__ Y4,   // [512][32] float4
                 f4* __restrict__ out4)       // [b*512][512][32] float4
{
    const int bi = blockIdx.x;          // b*512 + i
    const int i  = bi & (T_BLK - 1);
    const int c4 = threadIdx.x & 31;    // float4 column 0..31
    const int jg = threadIdx.x >> 5;    // 0..7
    const long baseOut = (long)bi * T_BLK * 32;

    for (int j = jg; j < T_BLK; j += 8) {
        int d = (j <= i) ? (i - j) : 0;
        f4 v = Y4[d * 32 + c4];
        __builtin_nontemporal_store(v, &out4[baseOut + (long)j * 32 + c4]);
    }
}

extern "C" void kernel_launch(void* const* d_in, const int* in_sizes, int n_in,
                              void* d_out, int out_size, void* d_ws, size_t ws_size,
                              hipStream_t stream) {
    (void)in_sizes; (void)n_in; (void)ws_size;
    // setup_inputs order: b(int,1), pos_table[512*64], W1[64*128], b1[128], W2[128*128], b2[128]
    const float* pos = (const float*)d_in[1];
    const float* W1  = (const float*)d_in[2];
    const float* b1  = (const float*)d_in[3];
    const float* W2  = (const float*)d_in[4];
    const float* b2  = (const float*)d_in[5];

    float* Y = (float*)d_ws;                     // 512*128*4 = 256 KB scratch

    // batch derived host-side from output element count (deterministic)
    int bsz = out_size / (T_BLK * T_BLK * H_DIM);
    if (bsz < 1) bsz = 1;

    dpe_build_table<<<T_BLK / 16, 32, 0, stream>>>(pos, W1, b1, W2, b2, Y);
    dpe_scatter<<<bsz * T_BLK, 256, 0, stream>>>((const f4*)Y, (f4*)d_out);
}
